// ModelWrapper_49907519980148
// MI455X (gfx1250) — matrix operand order repaired and checked by reading, not compiled, not run on hardware
//
#include <hip/hip_runtime.h>
#include <math.h>

typedef __bf16 bf16;
typedef __attribute__((ext_vector_type(16))) __bf16 v16bf;
typedef __attribute__((ext_vector_type(8)))  __bf16 v8bf;
typedef __attribute__((ext_vector_type(4)))  __bf16 v4bf;
typedef __attribute__((ext_vector_type(8)))  float  v8f;

#define Q_TOT       2048
#define D_K         512
#define N_TGT       100000
#define COLS_PER_WG 512
#define NSUB        8
#define SUBCOLS     64
#define LSTRIDE     520      // bf16 elements per staged row (pad: 1040B -> 4-bank shift/row)
#define NBLK        196      // ceil(100000 / 512)
#define KTOP        5

// bubble-up compare-swap for register-resident sorted top-5 (ascending)
#define CSWAP(da, db, ia, ib) \
  do { if ((db) < (da)) { float _td=(da); (da)=(db); (db)=_td; unsigned _ti=(ia); (ia)=(ib); (ib)=_ti; } } while (0)

__device__ __forceinline__ v16bf cat8(v8bf a, v8bf b) {
  return __builtin_shufflevector(a, b, 0,1,2,3,4,5,6,7,8,9,10,11,12,13,14,15);
}

// ---------------------------------------------------------------------------
// Kernel 1: norms + f32 -> bf16 hi/lo split of x.  One wave (32 lanes) per row,
// each lane owns 16 contiguous elements (4x float4).
// ---------------------------------------------------------------------------
__global__ __launch_bounds__(256) void knn_prep_kernel(
    const float* __restrict__ x, const float* __restrict__ T,
    bf16* __restrict__ xhi, bf16* __restrict__ xlo,
    float* __restrict__ x2, float* __restrict__ t2)
{
  const int wave = threadIdx.x >> 5;
  const int lane = threadIdx.x & 31;
  const int b    = blockIdx.x;

  if (b < Q_TOT / 8) {
    const int row = b * 8 + wave;
    const float* xr = x + (size_t)row * D_K;
    float s = 0.f;
#pragma unroll
    for (int u = 0; u < 16; u += 4) {
      float4 v = *(const float4*)(xr + lane * 16 + u);
      bf16 h0 = (bf16)v.x, h1 = (bf16)v.y, h2 = (bf16)v.z, h3 = (bf16)v.w;
      v4bf hv = {h0, h1, h2, h3};
      v4bf lv = {(bf16)(v.x - (float)h0), (bf16)(v.y - (float)h1),
                 (bf16)(v.z - (float)h2), (bf16)(v.w - (float)h3)};
      *(v4bf*)(xhi + (size_t)row * D_K + lane * 16 + u) = hv;
      *(v4bf*)(xlo + (size_t)row * D_K + lane * 16 + u) = lv;
      s += v.x * v.x + v.y * v.y + v.z * v.z + v.w * v.w;
    }
#pragma unroll
    for (int off = 16; off > 0; off >>= 1) s += __shfl_down(s, off, 32);
    if (lane == 0) x2[row] = s;
  } else {
    const int row = (b - Q_TOT / 8) * 8 + wave;
    if (row < N_TGT) {
      const float* tr = T + (size_t)row * D_K;
      float s = 0.f;
#pragma unroll
      for (int u = 0; u < 16; u += 4) {
        float4 v = *(const float4*)(tr + lane * 16 + u);
        s += v.x * v.x + v.y * v.y + v.z * v.z + v.w * v.w;
      }
#pragma unroll
      for (int off = 16; off > 0; off >>= 1) s += __shfl_down(s, off, 32);
      if (lane == 0) t2[row] = s;
    }
  }
}

// ---------------------------------------------------------------------------
// Kernel 2: per-N-chunk fused bf16-split WMMA GEMM + running top-5.
// Grid = 196 workgroups x 256 threads (8 waves). Each workgroup owns 512
// columns of T; stages 64-col x 512-K subtiles in LDS as bf16 hi/lo; every
// wave computes a 16(q) x 64(n) f32 tile with 12 v_wmma per 32-K chunk,
// interleaved across 4 accumulators to cover the bf16 WMMA hazard window.
// ---------------------------------------------------------------------------
__global__ __launch_bounds__(256) void knn_tile_kernel(
    const bf16* __restrict__ xhi, const bf16* __restrict__ xlo,
    const float* __restrict__ x2, const float* __restrict__ t2g,
    const float* __restrict__ T,
    float* __restrict__ pd, unsigned* __restrict__ pi)
{
  extern __shared__ char smem[];
  bf16*     sThi  = (bf16*)smem;                                   // 64*520*2   = 66560 B
  bf16*     sTlo  = sThi + SUBCOLS * LSTRIDE;                      //            = 66560 B
  float*    sDump = (float*)(smem + (size_t)2 * SUBCOLS * LSTRIDE * sizeof(bf16)); // 32768 B
  float*    sTopD = sDump + 8 * 16 * 64;                           // 2048*5*4   = 40960 B
  unsigned* sTopI = (unsigned*)(sTopD + Q_TOT * KTOP);             //            = 40960 B
  float*    sT2   = (float*)(sTopI + Q_TOT * KTOP);                // 64*4       =   256 B

  const int tid   = threadIdx.x;
  const int wave  = tid >> 5;
  const int lane  = tid & 31;
  const int blk   = blockIdx.x;
  const int ncol0 = blk * COLS_PER_WG;

  for (int i = tid; i < Q_TOT * KTOP; i += 256) { sTopD[i] = 3.0e38f; sTopI[i] = 0u; }
  __syncthreads();

  for (int sub = 0; sub < NSUB; ++sub) {
    const int colbase = ncol0 + sub * SUBCOLS;

    if (tid < SUBCOLS) {
      int gc = colbase + tid;
      sT2[tid] = (gc < N_TGT) ? t2g[gc] : 3.0e38f;
    }
    // stage 64 cols x 512 K of T: f32 global -> bf16 hi/lo LDS
    for (int idx = tid * 4; idx < SUBCOLS * D_K; idx += 256 * 4) {
      int c = idx >> 9;              // / 512
      int k = idx & (D_K - 1);
      int gcol = colbase + c;
      float4 v = make_float4(0.f, 0.f, 0.f, 0.f);
      if (gcol < N_TGT) {
        v = *(const float4*)(T + (size_t)gcol * D_K + k);
        if (gcol + SUBCOLS < N_TGT)   // global_prefetch_b8 next subtile
          __builtin_prefetch(T + (size_t)(gcol + SUBCOLS) * D_K + k, 0, 1);
      }
      bf16 h0 = (bf16)v.x, h1 = (bf16)v.y, h2 = (bf16)v.z, h3 = (bf16)v.w;
      v4bf hv = {h0, h1, h2, h3};
      v4bf lv = {(bf16)(v.x - (float)h0), (bf16)(v.y - (float)h1),
                 (bf16)(v.z - (float)h2), (bf16)(v.w - (float)h3)};
      *(v4bf*)(sThi + (size_t)c * LSTRIDE + k) = hv;
      *(v4bf*)(sTlo + (size_t)c * LSTRIDE + k) = lv;
    }
    __syncthreads();

    for (int qiter = 0; qiter < Q_TOT / 128; ++qiter) {
      const int qbase = qiter * 128 + wave * 16;
      const int mrow  = qbase + (lane & 15);
      const int kaoff = (lane >> 4) * 8;    // A: lanes 16-31 hold K+8 block
      const int kboff = (lane >> 4) * 8;    // B: lanes 16-31 hold K+8 block
      const int bn    = lane & 15;
      const bf16* xh  = xhi + (size_t)mrow * D_K;
      const bf16* xl  = xlo + (size_t)mrow * D_K;

      v8f acc[4] = {};
      for (int kc = 0; kc < D_K; kc += 32) {
        // A fragments (16x32): per lane two contiguous 16B row segments
        v16bf Ah = cat8(*(const v8bf*)(xh + kc + kaoff),
                        *(const v8bf*)(xh + kc + kaoff + 16));
        v16bf Al = cat8(*(const v8bf*)(xl + kc + kaoff),
                        *(const v8bf*)(xl + kc + kaoff + 16));

        // load all 8 B fragments up front so WMMAs can interleave accumulators
        v16bf Bh[4], Bl[4];
#pragma unroll
        for (int ns = 0; ns < 4; ++ns) {
          const bf16* bp = sThi + (size_t)(ns * 16 + bn) * LSTRIDE + kc + kboff;
          const bf16* bq = sTlo + (size_t)(ns * 16 + bn) * LSTRIDE + kc + kboff;
          Bh[ns] = cat8(*(const v8bf*)bp, *(const v8bf*)(bp + 16));
          Bl[ns] = cat8(*(const v8bf*)bq, *(const v8bf*)(bq + 16));
        }
        // 12 WMMAs grouped by product: each accumulator's reuse distance is 4,
        // covering the bf16 WMMA->WMMA hazard window (1 NOP + 4 coexec) without
        // compiler-inserted v_nop padding.
#pragma unroll
        for (int ns = 0; ns < 4; ++ns)
          acc[ns] = __builtin_amdgcn_wmma_f32_16x16x32_bf16(false, Ah, false, Bh[ns],
                                                            (short)0, acc[ns], false, false);
#pragma unroll
        for (int ns = 0; ns < 4; ++ns)
          acc[ns] = __builtin_amdgcn_wmma_f32_16x16x32_bf16(false, Ah, false, Bl[ns],
                                                            (short)0, acc[ns], false, false);
#pragma unroll
        for (int ns = 0; ns < 4; ++ns)
          acc[ns] = __builtin_amdgcn_wmma_f32_16x16x32_bf16(false, Al, false, Bh[ns],
                                                            (short)0, acc[ns], false, false);
      }

      // dump C tiles: VGPR r -> row r (lanes<16) / r+8 (lanes>=16), col = lane%16
      float* dumpw = sDump + wave * (16 * 64);
      const int mhalf = (lane >> 4) * 8;
#pragma unroll
      for (int ns = 0; ns < 4; ++ns)
#pragma unroll
        for (int r = 0; r < 8; ++r)
          dumpw[(mhalf + r) * 64 + ns * 16 + bn] = acc[ns][r];
      __syncthreads();

      // serial top-5 scan: one thread per query row, merge into LDS running top-5
      if (tid < 128) {
        const int q = qiter * 128 + tid;
        const float* row = sDump + tid * 64;
        const float xq = x2[q];
        float    bd0 = sTopD[q*5+0], bd1 = sTopD[q*5+1], bd2 = sTopD[q*5+2],
                 bd3 = sTopD[q*5+3], bd4 = sTopD[q*5+4];
        unsigned bi0 = sTopI[q*5+0], bi1 = sTopI[q*5+1], bi2 = sTopI[q*5+2],
                 bi3 = sTopI[q*5+3], bi4 = sTopI[q*5+4];
        for (int c = 0; c < SUBCOLS; ++c) {
          float d2v = xq + sT2[c] - 2.0f * row[c];
          if (d2v < bd4) {
            bd4 = d2v; bi4 = (unsigned)(colbase + c);
            CSWAP(bd3, bd4, bi3, bi4);
            CSWAP(bd2, bd3, bi2, bi3);
            CSWAP(bd1, bd2, bi1, bi2);
            CSWAP(bd0, bd1, bi0, bi1);
          }
        }
        sTopD[q*5+0] = bd0; sTopD[q*5+1] = bd1; sTopD[q*5+2] = bd2;
        sTopD[q*5+3] = bd3; sTopD[q*5+4] = bd4;
        sTopI[q*5+0] = bi0; sTopI[q*5+1] = bi1; sTopI[q*5+2] = bi2;
        sTopI[q*5+3] = bi3; sTopI[q*5+4] = bi4;
      }
      __syncthreads();
    }
  }

  // flush per-block partial top-5 to workspace: [q][blk][5]
  for (int i = tid; i < Q_TOT * KTOP; i += 256) {
    int q = i / KTOP, j = i % KTOP;
    pd[(size_t)q * NBLK * KTOP + (size_t)blk * KTOP + j] = sTopD[i];
    pi[(size_t)q * NBLK * KTOP + (size_t)blk * KTOP + j] = sTopI[i];
  }
}

// ---------------------------------------------------------------------------
// Kernel 3: merge 196 partial top-5 lists per query; emit dists, indices,
// transposed labels (all as f32, concatenated in reference return order).
// ---------------------------------------------------------------------------
__global__ __launch_bounds__(256) void knn_merge_kernel(
    const float* __restrict__ pd, const unsigned* __restrict__ pi,
    const int* __restrict__ labels, float* __restrict__ out)
{
  const int q = blockIdx.x * blockDim.x + threadIdx.x;
  if (q >= Q_TOT) return;

  float    bd0 = 3.0e38f, bd1 = 3.0e38f, bd2 = 3.0e38f, bd3 = 3.0e38f, bd4 = 3.0e38f;
  unsigned bi0 = 0, bi1 = 0, bi2 = 0, bi3 = 0, bi4 = 0;
  const float*    rd = pd + (size_t)q * NBLK * KTOP;
  const unsigned* ri = pi + (size_t)q * NBLK * KTOP;
  for (int c = 0; c < NBLK * KTOP; ++c) {
    float d2v = rd[c];
    if (d2v < bd4) {
      bd4 = d2v; bi4 = ri[c];
      CSWAP(bd3, bd4, bi3, bi4);
      CSWAP(bd2, bd3, bi2, bi3);
      CSWAP(bd1, bd2, bi1, bi2);
      CSWAP(bd0, bd1, bi0, bi1);
    }
  }

  float    bd[KTOP] = {bd0, bd1, bd2, bd3, bd4};
  unsigned bi[KTOP] = {bi0, bi1, bi2, bi3, bi4};
#pragma unroll
  for (int j = 0; j < KTOP; ++j) {
    unsigned gi = (bi[j] < N_TGT) ? bi[j] : 0u;
    out[(size_t)q * KTOP + j]                         = sqrtf(fmaxf(bd[j], 0.f)); // topk_dists [Q,5]
    out[(size_t)Q_TOT * KTOP + (size_t)q * KTOP + j]  = (float)gi;                // topk_inds  [Q,5]
    out[(size_t)2 * Q_TOT * KTOP + (size_t)j * Q_TOT + q] = (float)labels[gi];    // pred       [5,Q]
  }
}

// ---------------------------------------------------------------------------
extern "C" void kernel_launch(void* const* d_in, const int* in_sizes, int n_in,
                              void* d_out, int out_size, void* d_ws, size_t ws_size,
                              hipStream_t stream) {
  const float* x      = (const float*)d_in[0];
  const float* T      = (const float*)d_in[1];
  const int*   labels = (const int*)d_in[2];
  (void)in_sizes; (void)n_in; (void)out_size; (void)ws_size;

  char* ws = (char*)d_ws;
  size_t off = 0;
  bf16*  xhi = (bf16*)(ws + off);  off += (size_t)Q_TOT * D_K * sizeof(bf16);   // 2 MB
  bf16*  xlo = (bf16*)(ws + off);  off += (size_t)Q_TOT * D_K * sizeof(bf16);   // 2 MB
  float* x2  = (float*)(ws + off); off += (size_t)Q_TOT * sizeof(float);
  float* t2  = (float*)(ws + off); off += (size_t)((N_TGT + 127) & ~127u) * sizeof(float);
  float* pd  = (float*)(ws + off); off += (size_t)Q_TOT * NBLK * KTOP * sizeof(float);     // 8 MB
  unsigned* pi = (unsigned*)(ws + off); off += (size_t)Q_TOT * NBLK * KTOP * sizeof(unsigned); // 8 MB

  knn_prep_kernel<<<dim3(Q_TOT / 8 + (N_TGT + 7) / 8), dim3(256), 0, stream>>>(
      x, T, xhi, xlo, x2, t2);

  const size_t smem = (size_t)2 * SUBCOLS * LSTRIDE * sizeof(bf16)  // T hi/lo stage
                    + (size_t)8 * 16 * 64 * sizeof(float)           // C dump
                    + (size_t)Q_TOT * KTOP * (sizeof(float) + sizeof(unsigned)) // running top-5
                    + (size_t)SUBCOLS * sizeof(float);              // t2 tile
  knn_tile_kernel<<<dim3(NBLK), dim3(256), smem, stream>>>(
      xhi, xlo, x2, t2, T, pd, pi);

  knn_merge_kernel<<<dim3((Q_TOT + 255) / 256), dim3(256), 0, stream>>>(
      pd, pi, labels, (float*)d_out);
}
